// ForceOnlyDynamicsDecoderRNN_61899068670239
// MI455X (gfx1250) — compile-verified, matrix operand order checked by reading, not run on hardware
//
#include <hip/hip_runtime.h>

// ---------------------------------------------------------------------------
// GRU persistent kernel for MI455X (gfx1250, wave32, WMMA bf16 16x16x32)
//
// B=8192, T=512, H=512, gates G=3H=1536, input=[f(2), y(4), z(64)]=70, OUT=4
//
// Strategy:
//   * precompute gz_const[b][g] = z_dyn@Wz.T + b_ih (+ b_hh for r/z gates)
//   * pack w_hh.T into bf16 WMMA-B fragment layout (lane-major 32B chunks)
//   * persistent kernel: 128 WGs x 256 thr; each WG owns 64 batch rows,
//     keeps h fp32 + double-buffered h bf16 in LDS for all 512 steps,
//     recurrent GEMM via v_wmma_f32_16x16x32_bf16 (B-frag reused over 4 mb)
//   * w_out/b_out staged in LDS once (loop-invariant output head)
// ---------------------------------------------------------------------------

#define B_    8192
#define T_    512
#define H_    512
#define G_    1536
#define ZD    64
#define INTOT 70
#define OUTD  4
#define BTILE 64          // batch rows per workgroup
#define HBS   520         // padded row stride (elements) for LDS h buffers

typedef __attribute__((ext_vector_type(16))) __bf16       v16bf;
typedef __attribute__((ext_vector_type(8)))  float        v8f;
typedef __attribute__((ext_vector_type(4)))  unsigned int uint4v;

union Frag16 { uint4v q[2]; v16bf v; };

__device__ __forceinline__ float sigm_(float x) { return 1.0f / (1.0f + __expf(-x)); }
__device__ __forceinline__ float tanh_(float x) { return 1.0f - 2.0f / (1.0f + __expf(2.0f * x)); }

// ---------------------------------------------------------------------------
// Pack w_hh (1536x512 row-major fp32) into bf16 WMMA B-fragments.
// Tile (ntile in [0,96), kt in [0,16)): 32 lanes x 16 bf16 slots (32B/lane).
// Layout mirrors the documented 16-bit A layout: lane%16 = N, lane/16 = half,
// slot s -> vgpr v=s/2, elem e=s&1, k_local = 2*(v&3) + e + 8*half + 16*(v/4)
// ---------------------------------------------------------------------------
extern "C" __global__ __launch_bounds__(256)
void pack_whh_kernel(const float* __restrict__ whh, unsigned short* __restrict__ pk) {
    int idx = blockIdx.x * 256 + threadIdx.x;           // < 96*16*32*16 = 786432
    int slot  = idx & 15;
    int lane  = (idx >> 4) & 31;
    int kt    = (idx >> 9) & 15;
    int ntile = idx >> 13;
    int nsub = lane & 15, half = lane >> 4;
    int v = slot >> 1, e = slot & 1;
    int klocal = 2 * (v & 3) + e + 8 * half + 16 * (v >> 2);
    int n = ntile * 16 + nsub;                          // gate-output index g
    int k = kt * 32 + klocal;                           // hidden index
    __bf16 b = (__bf16)whh[n * H_ + k];
    pk[idx] = __builtin_bit_cast(unsigned short, b);
}

// ---------------------------------------------------------------------------
// gz_const[b][g] = sum_c z_dyn[b][c]*w_ih[g][6+c] + b_ih[g] + (g<2H ? b_hh[g] : 0)
// (b_hh for the n-gate must stay inside r*h_n, so it is NOT folded there)
// ---------------------------------------------------------------------------
extern "C" __global__ __launch_bounds__(256)
void gz_const_kernel(const float* __restrict__ z_dyn, const float* __restrict__ w_ih,
                     const float* __restrict__ b_ih, const float* __restrict__ b_hh,
                     float* __restrict__ gz) {
    size_t idx = (size_t)blockIdx.x * 256 + threadIdx.x;    // B_*G_ total, exact
    int g   = (int)(idx % G_);
    int row = (int)(idx / G_);
    float s = b_ih[g] + (g < 2 * H_ ? b_hh[g] : 0.0f);
    const float* zr = z_dyn + (size_t)row * ZD;
    const float* wg = w_ih + (size_t)g * INTOT + 6;
    #pragma unroll 8
    for (int c = 0; c < ZD; ++c) s = fmaf(zr[c], wg[c], s);
    gz[idx] = s;
}

// ---------------------------------------------------------------------------
// Persistent GRU kernel. Grid: 128 x 256 threads (8 waves), dyn LDS 275984 B.
// LDS map:  [0)            h_bf16 buf0   64x520 bf16  = 66560 B
//           [66560)        h_bf16 buf1                = 66560 B
//           [133120)       h_f32         64x520 f32   = 133120 B
//           [266240)       f_t stage     64x2  f32    = 512 B
//           [266752)       y_prev        64x4  f32    = 1024 B
//           [267776)       w_out         4x512 f32    = 8192 B
//           [275968)       b_out         4     f32    = 16 B
// ---------------------------------------------------------------------------
extern "C" __global__ __launch_bounds__(256)
void gru_persist_kernel(const float* __restrict__ f_seq, const float* __restrict__ init_y,
                        const float* __restrict__ w_ih,  const float* __restrict__ b_hh,
                        const float* __restrict__ w_out, const float* __restrict__ b_out,
                        const float* __restrict__ gz,    const unsigned short* __restrict__ pk,
                        float* __restrict__ out) {
    extern __shared__ char smem[];
    unsigned short* hb0 = (unsigned short*)(smem);
    unsigned short* hb1 = (unsigned short*)(smem + 66560);
    float*          hf  = (float*)(smem + 133120);
    float*          fb  = (float*)(smem + 266240);
    float*          yb  = (float*)(smem + 266752);
    float*          wos = (float*)(smem + 267776);
    float*          bos = (float*)(smem + 275968);

    const int tid  = threadIdx.x;
    const int lane = tid & 31;
    const int wv   = tid >> 5;
    const int nsub = lane & 15;
    const int half = lane >> 4;
    const int row0 = blockIdx.x * BTILE;

    // ---- init: h = 0 (fp32 + bf16 buf0), y_prev = init_y, stage w_out/b_out
    for (int i = tid; i < BTILE * HBS; i += 256) { hb0[i] = 0; hf[i] = 0.0f; }
    { int r = tid >> 2, o = tid & 3;
      yb[tid] = init_y[(size_t)(row0 + r) * OUTD + o]; }
    #pragma unroll
    for (int i = 0; i < OUTD * H_ / 256; ++i) {
        int k = i * 256 + tid;
        wos[k] = w_out[k];
    }
    if (tid < OUTD) bos[tid] = b_out[tid];
    __syncthreads();

    #pragma unroll 1
    for (int t = 0; t < T_; ++t) {
        const unsigned short* hcur = (t & 1) ? hb1 : hb0;
        unsigned short*       hnxt = (t & 1) ? hb0 : hb1;

        // ---- stage f_t for this step ----
        if (tid < BTILE * 2) {
            int r = tid >> 1, c = tid & 1;
            fb[tid] = f_seq[((size_t)(row0 + r) * T_ + t) * 2 + c];
        }
        __syncthreads();

        // ---- recurrent GEMM (WMMA) + gate math ----
        #pragma unroll 1
        for (int jj = 0; jj < 4; ++jj) {
            const int jb = wv * 4 + jj;          // column tile in [0,32)
            const int j  = jb * 16 + nsub;       // hidden column this lane owns

            v8f acc[4][3];
            #pragma unroll
            for (int mb = 0; mb < 4; ++mb)
                #pragma unroll
                for (int g = 0; g < 3; ++g) acc[mb][g] = (v8f){};

            #pragma unroll 2
            for (int kt = 0; kt < 16; ++kt) {
                // B fragments for the r/z/n gate column tiles (reused over mb)
                Frag16 bf[3];
                #pragma unroll
                for (int g = 0; g < 3; ++g) {
                    const int ntile = g * 32 + jb;
                    const uint4v* p = (const uint4v*)
                        (pk + ((size_t)(ntile * 16 + kt) * 32 + lane) * 16);
                    bf[g].q[0] = p[0];
                    bf[g].q[1] = p[1];
                }
                #pragma unroll
                for (int mb = 0; mb < 4; ++mb) {
                    const int arow = mb * 16 + nsub;  // A-matrix M index
                    const uint4v* pa = (const uint4v*)
                        (hcur + arow * HBS + kt * 32 + half * 8);
                    Frag16 af;
                    af.q[0] = pa[0];
                    af.q[1] = pa[2];                  // +32 bytes
                    #pragma unroll
                    for (int g = 0; g < 3; ++g)
                        acc[mb][g] = __builtin_amdgcn_wmma_f32_16x16x32_bf16(
                            false, af.v, false, bf[g].v, (short)0, acc[mb][g],
                            false, false);
                }
            }

            // per-lane gate-column weights for the rank-6 dynamic gi part
            float wd[3][6];
            #pragma unroll
            for (int g = 0; g < 3; ++g) {
                const float* wrow = w_ih + (size_t)(g * H_ + j) * INTOT;
                #pragma unroll
                for (int c = 0; c < 6; ++c) wd[g][c] = wrow[c];
            }
            const float bhn = b_hh[2 * H_ + j];

            #pragma unroll 1
            for (int mb = 0; mb < 4; ++mb) {
                #pragma unroll
                for (int v = 0; v < 8; ++v) {
                    const int row = mb * 16 + v + half * 8;   // C/D layout
                    const size_t rg = (size_t)(row0 + row) * G_;
                    const float f0 = fb[row * 2], f1 = fb[row * 2 + 1];
                    const float y0 = yb[row * 4],     y1 = yb[row * 4 + 1];
                    const float y2 = yb[row * 4 + 2], y3 = yb[row * 4 + 3];

                    float pr = gz[rg + j] + acc[mb][0][v];
                    pr += wd[0][0]*f0 + wd[0][1]*f1 + wd[0][2]*y0 + wd[0][3]*y1
                        + wd[0][4]*y2 + wd[0][5]*y3;
                    float pz = gz[rg + H_ + j] + acc[mb][1][v];
                    pz += wd[1][0]*f0 + wd[1][1]*f1 + wd[1][2]*y0 + wd[1][3]*y1
                        + wd[1][4]*y2 + wd[1][5]*y3;
                    float pn = gz[rg + 2 * H_ + j];
                    pn += wd[2][0]*f0 + wd[2][1]*f1 + wd[2][2]*y0 + wd[2][3]*y1
                        + wd[2][4]*y2 + wd[2][5]*y3;

                    const float r  = sigm_(pr);
                    const float zz = sigm_(pz);
                    const float hn = acc[mb][2][v] + bhn;     // gh_n + b_hh_n
                    const float n  = tanh_(pn + r * hn);
                    const float hold = hf[row * HBS + j];
                    const float hnew = (1.0f - zz) * n + zz * hold;

                    hf[row * HBS + j] = hnew;
                    __bf16 hb = (__bf16)hnew;
                    hnxt[row * HBS + j] = __builtin_bit_cast(unsigned short, hb);
                }
            }
        }
        __syncthreads();

        // ---- output head: y = h_new @ w_out.T + b_out  (one (row,out)/thread)
        {
            const int r = tid >> 2, o = tid & 3;
            float s = bos[o];
            const float* hrow = hf + r * HBS;
            const float* wo   = wos + o * H_;
            #pragma unroll 4
            for (int k = 0; k < H_; k += 4) {
                const float4 hv  = *(const float4*)(hrow + k);
                const float4 wv4 = *(const float4*)(wo + k);
                s = fmaf(hv.x, wv4.x, s); s = fmaf(hv.y, wv4.y, s);
                s = fmaf(hv.z, wv4.z, s); s = fmaf(hv.w, wv4.w, s);
            }
            yb[tid] = s;                                       // feeds next step
            out[((size_t)(row0 + r) * T_ + t) * OUTD + o] = s;
        }
        __syncthreads();
    }
}

// ---------------------------------------------------------------------------
extern "C" void kernel_launch(void* const* d_in, const int* in_sizes, int n_in,
                              void* d_out, int out_size, void* d_ws, size_t ws_size,
                              hipStream_t stream) {
    const float* f_seq  = (const float*)d_in[0];
    const float* init_y = (const float*)d_in[1];
    const float* z_dyn  = (const float*)d_in[2];
    const float* w_ih   = (const float*)d_in[3];
    const float* w_hh   = (const float*)d_in[4];
    const float* b_ih   = (const float*)d_in[5];
    const float* b_hh   = (const float*)d_in[6];
    const float* w_out  = (const float*)d_in[7];
    const float* b_out  = (const float*)d_in[8];
    float* out = (float*)d_out;

    // workspace: gz_const (B*G fp32 = 48 MiB) | packed w_hh bf16 (1.5 MiB)
    float*          gzc = (float*)d_ws;
    unsigned short* pkw = (unsigned short*)((char*)d_ws + (size_t)B_ * G_ * sizeof(float));

    pack_whh_kernel<<<(G_ * H_) / 256, 256, 0, stream>>>(w_hh, pkw);
    gz_const_kernel<<<((size_t)B_ * G_) / 256, 256, 0, stream>>>(z_dyn, w_ih, b_ih, b_hh, gzc);

    const size_t lds_bytes = 275984;  // 2*66560 + 133120 + 512 + 1024 + 8192 + 16
    gru_persist_kernel<<<B_ / BTILE, 256, lds_bytes, stream>>>(
        f_seq, init_y, w_ih, b_hh, w_out, b_out, gzc, pkw, out);
}
